// ModelManager_71107478552913
// MI455X (gfx1250) — matrix operand order repaired and checked by reading, not compiled, not run on hardware
//
#include <hip/hip_runtime.h>
#include <hip/hip_bf16.h>
#include <cstdint>

#define B_  64
#define T_  512
#define D_  1024
#define H_  1024
#define E_  128
#define O_  512
#define KZ  (D_ + E_)   /* 1152 : concat(x, prev_emb) */
#define G4  (4 * H_)    /* 4096 : i,f,g,o gate rows   */

typedef __attribute__((ext_vector_type(16))) __bf16        bf16x16;
typedef __attribute__((ext_vector_type(8)))  float          floatx8;
typedef __attribute__((ext_vector_type(8)))  unsigned short ushortx8;

union BFVec { bf16x16 bf; ushortx8 us[2]; };

__device__ __forceinline__ unsigned short f2bf(float f) {
  union { float f; unsigned u; } v; v.f = f;
  unsigned r = v.u + 0x7FFFu + ((v.u >> 16) & 1u);   // round-to-nearest-even
  return (unsigned short)(r >> 16);
}

__device__ __forceinline__ float sigm(float x) { return 1.0f / (1.0f + __expf(-x)); }

// ---------------------------------------------------------------------------
// Core WMMA phase: one wave computes a 32-row x 64-batch tile against K cols.
// Ping-pong double buffering: while WMMAs consume buffer p, the 12 loads for
// buffer 1-p are already in flight -> partial s_wait_loadcnt instead of full
// drains. K must be a multiple of 64 (1152 and 1024 both are).
// Operand layouts match ISA 7.12.2 natively:
//   A lane L: row L%16, K-chunks at (L/16)*8 and 16+(L/16)*8  (2x b128 loads)
//   B lane L: col L%16, K-chunks at (L/16)*8 and 16+(L/16)*8  (2x b128 loads)
// ---------------------------------------------------------------------------
template<int K>
__device__ __forceinline__ void mma_phase(const unsigned short* __restrict__ W,
                                          const unsigned short* __restrict__ act,
                                          int rowBase, int laneM, int laneHalf,
                                          floatx8* acc)
{
  const unsigned short* aRow0 = W + (size_t)(rowBase + laneM) * K + laneHalf * 8;
  const unsigned short* aRow1 = aRow0 + (size_t)16 * K;
  const unsigned short* bBase = act + (size_t)laneM * K + laneHalf * 8;

  BFVec a0[2], a1[2], bv[2][4];

  auto loadA = [&](const unsigned short* p, int k, BFVec& v) {
    v.us[0] = *(const ushortx8*)(p + k);
    v.us[1] = *(const ushortx8*)(p + k + 16);
  };
  auto loadB = [&](int n, int k, BFVec& v) {
    const unsigned short* p = bBase + (size_t)n * 16 * K + k;
    v.us[0] = *(const ushortx8*)(p);
    v.us[1] = *(const ushortx8*)(p + 16);
  };

  // preload buffer 0 (k = 0)
  loadA(aRow0, 0, a0[0]);
  loadA(aRow1, 0, a1[0]);
#pragma unroll
  for (int n = 0; n < 4; ++n) loadB(n, 0, bv[0][n]);

  for (int k = 0; k < K; k += 64) {
    const int k1 = (k + 32 < K) ? k + 32 : 0;   // clamped dummy prefetch at tail
    loadA(aRow0, k1, a0[1]);
    loadA(aRow1, k1, a1[1]);
#pragma unroll
    for (int n = 0; n < 4; ++n) loadB(n, k1, bv[1][n]);
#pragma unroll
    for (int n = 0; n < 4; ++n) {
      acc[n]     = __builtin_amdgcn_wmma_f32_16x16x32_bf16(
                       false, a0[0].bf, false, bv[0][n].bf, (short)0, acc[n],     false, false);
      acc[4 + n] = __builtin_amdgcn_wmma_f32_16x16x32_bf16(
                       false, a1[0].bf, false, bv[0][n].bf, (short)0, acc[4 + n], false, false);
    }
    const int k2 = (k + 64 < K) ? k + 64 : 0;
    loadA(aRow0, k2, a0[0]);
    loadA(aRow1, k2, a1[0]);
#pragma unroll
    for (int n = 0; n < 4; ++n) loadB(n, k2, bv[0][n]);
#pragma unroll
    for (int n = 0; n < 4; ++n) {
      acc[n]     = __builtin_amdgcn_wmma_f32_16x16x32_bf16(
                       false, a0[1].bf, false, bv[1][n].bf, (short)0, acc[n],     false, false);
      acc[4 + n] = __builtin_amdgcn_wmma_f32_16x16x32_bf16(
                       false, a1[1].bf, false, bv[1][n].bf, (short)0, acc[4 + n], false, false);
    }
  }
}

// ---------------------------------------------------------------- init / convert
__global__ void f2bf_kernel(const float* __restrict__ src,
                            unsigned short* __restrict__ dst, int n) {
  int i = blockIdx.x * blockDim.x + threadIdx.x;
  if (i < n) dst[i] = f2bf(src[i]);
}

__global__ void init_state_kernel(const float* __restrict__ b_ih,
                                  const float* __restrict__ b_hh,
                                  const float* __restrict__ init_t,
                                  float* __restrict__ bias_g,         // [4096]
                                  unsigned short* __restrict__ h_bf,  // [B][H]
                                  float* __restrict__ c_f,            // [H][B] (j*64+b)
                                  unsigned short* __restrict__ pemb)  // [B][E]
{
  int i = blockIdx.x * blockDim.x + threadIdx.x;     // 65536 threads
  if (i < B_ * H_) { h_bf[i] = 0; c_f[i] = 0.0f; }
  if (i < G4)       bias_g[i] = b_ih[i] + b_hh[i];
  if (i < B_ * E_)  pemb[i] = f2bf(init_t[i & (E_ - 1)]);
}

// ---------------------------------------------------------------- per-step: build z
__global__ void zbuild_kernel(const float* __restrict__ enc,
                              const unsigned short* __restrict__ pemb,
                              unsigned short* __restrict__ z, int t) {
  int e = blockIdx.x * blockDim.x + threadIdx.x;     // 64*1152
  if (e >= B_ * KZ) return;
  int b = e / KZ, k = e - b * KZ;
  unsigned short v;
  if (k < D_) v = f2bf(enc[((size_t)b * T_ + t) * D_ + k]);
  else        v = pemb[b * E_ + (k - D_)];
  z[e] = v;
}

// ---------------------------------------------------------------- gates GEMM (WMMA)
// gates[row(4096)][batch(64)] = Wih·z^T + Whh·h^T + bias
// grid: 32 blocks x 128 threads; wave owns 32 rows x full 64-batch.
__global__ __launch_bounds__(128) void gates_kernel(
    const unsigned short* __restrict__ Wih,   // [4096][1152] bf16 row-major
    const unsigned short* __restrict__ Whh,   // [4096][1024] bf16 row-major
    const unsigned short* __restrict__ z,     // [64][1152]   bf16 row-major
    const unsigned short* __restrict__ h,     // [64][1024]   bf16 row-major
    const float* __restrict__ bias,           // [4096]
    float* __restrict__ gates)                // [4096][64]
{
  const int lane     = threadIdx.x & 31;
  const int wave     = threadIdx.x >> 5;      // 4 waves
  const int laneM    = lane & 15;
  const int laneHalf = lane >> 4;
  const int rowBase  = blockIdx.x * 128 + wave * 32;

  floatx8 acc[8] = {};                        // 2 M-tiles x 4 N-tiles
  mma_phase<KZ>(Wih, z, rowBase, laneM, laneHalf, acc);
  mma_phase<H_>(Whh, h, rowBase, laneM, laneHalf, acc);

  // D layout: lane L, vgpr r holds D[r + 8*(L/16)][L%16]
#pragma unroll
  for (int m = 0; m < 2; ++m)
#pragma unroll
    for (int r = 0; r < 8; ++r) {
      int row = rowBase + m * 16 + r + 8 * laneHalf;
      float bs = bias[row];
#pragma unroll
      for (int n = 0; n < 4; ++n)
        gates[(size_t)row * B_ + n * 16 + laneM] = acc[m * 4 + n][r] + bs;
    }
}

// ---------------------------------------------------------------- LSTM cell (+mask)
__global__ void cell_kernel(const float* __restrict__ gates,   // [4096][64]
                            float* __restrict__ c,             // [H][B] j*64+b
                            unsigned short* __restrict__ h_bf, // [B][H]
                            const int* __restrict__ seq_lens, int t) {
  int e = blockIdx.x * blockDim.x + threadIdx.x;   // 65536
  int b = e & (B_ - 1), j = e >> 6;
  float iv = sigm(gates[(size_t)(j          ) * B_ + b]);
  float fv = sigm(gates[(size_t)(j +     H_ ) * B_ + b]);
  float gv = tanhf(gates[(size_t)(j + 2 * H_) * B_ + b]);
  float ov = sigm(gates[(size_t)(j + 3 * H_) * B_ + b]);
  float cold = c[j * B_ + b];
  float cnew = fv * cold + iv * gv;
  float hnew = ov * tanhf(cnew);
  if (t < seq_lens[b]) {                  // ragged-length freeze
    c[j * B_ + b]    = cnew;
    h_bf[b * H_ + j] = f2bf(hnew);
  }
}

// ---------------------------------------------------------------- logits GEMM (WMMA)
// grid: 4 blocks x 128 threads; wave owns 32 rows x 64 batch, K = 1024.
__global__ __launch_bounds__(128) void logits_kernel(
    const unsigned short* __restrict__ Wlin,  // [512][1024] bf16
    const unsigned short* __restrict__ h,     // [64][1024]  bf16
    const float* __restrict__ b_lin,          // [512]
    float* __restrict__ logits)               // [512][64]
{
  const int lane     = threadIdx.x & 31;
  const int wave     = threadIdx.x >> 5;
  const int laneM    = lane & 15;
  const int laneHalf = lane >> 4;
  const int rowBase  = blockIdx.x * 128 + wave * 32;

  floatx8 acc[8] = {};
  mma_phase<H_>(Wlin, h, rowBase, laneM, laneHalf, acc);

#pragma unroll
  for (int m = 0; m < 2; ++m)
#pragma unroll
    for (int r = 0; r < 8; ++r) {
      int row = rowBase + m * 16 + r + 8 * laneHalf;
      float bs = b_lin[row];
#pragma unroll
      for (int n = 0; n < 4; ++n)
        logits[(size_t)row * B_ + n * 16 + laneM] = acc[m * 4 + n][r] + bs;
    }
}

// ---------------------------------------------------------------- argmax + emb + out
__global__ __launch_bounds__(128) void out_kernel(
    const float* __restrict__ logits,     // [512][64]
    const float* __restrict__ emb,        // [512][128]
    const int* __restrict__ seq_lens,
    unsigned short* __restrict__ pemb,    // [64][128] bf16
    float* __restrict__ out, int t)       // [64][512][512]
{
  __shared__ float sv[128];
  __shared__ int   si[128];
  int b = blockIdx.x, tid = threadIdx.x;
  bool valid = t < seq_lens[b];

  float best = -3.402823466e+38f; int bi = 0;
  for (int r = tid; r < O_; r += 128) {
    float v = logits[(size_t)r * B_ + b];
    if (v > best) { best = v; bi = r; }        // rows ascend -> first max kept
  }
  sv[tid] = best; si[tid] = bi;
  __syncthreads();
  for (int s = 64; s > 0; s >>= 1) {
    if (tid < s) {
      float v2 = sv[tid + s]; int i2 = si[tid + s];
      if (v2 > sv[tid] || (v2 == sv[tid] && i2 < si[tid])) { sv[tid] = v2; si[tid] = i2; }
    }
    __syncthreads();
  }
  int idx = si[0];

  float* ob = out + ((size_t)b * T_ + t) * O_;
  for (int o = tid; o < O_; o += 128)
    ob[o] = valid ? logits[(size_t)o * B_ + b] : 0.0f;
  if (valid && tid < E_)
    pemb[b * E_ + tid] = f2bf(emb[(size_t)idx * E_ + tid]);
}

// ---------------------------------------------------------------- host
extern "C" void kernel_launch(void* const* d_in, const int* in_sizes, int n_in,
                              void* d_out, int out_size, void* d_ws, size_t ws_size,
                              hipStream_t stream) {
  const float* enc      = (const float*)d_in[0];
  const int*   seq_lens = (const int*)  d_in[1];
  const float* W_ih     = (const float*)d_in[2];
  const float* W_hh     = (const float*)d_in[3];
  const float* b_ih     = (const float*)d_in[4];
  const float* b_hh     = (const float*)d_in[5];
  const float* W_lin    = (const float*)d_in[6];
  const float* b_lin    = (const float*)d_in[7];
  const float* emb      = (const float*)d_in[8];
  const float* init_t   = (const float*)d_in[9];
  float* out = (float*)d_out;

  char* ws = (char*)d_ws;
  size_t off = 0;
  auto alloc = [&](size_t bytes) -> void* {
    void* p = ws + off;
    off += (bytes + 255) & ~(size_t)255;
    return p;
  };
  unsigned short* Wih_bf  = (unsigned short*)alloc((size_t)G4 * KZ * 2);
  unsigned short* Whh_bf  = (unsigned short*)alloc((size_t)G4 * H_ * 2);
  unsigned short* Wlin_bf = (unsigned short*)alloc((size_t)O_ * H_ * 2);
  float*          bias_g  = (float*)         alloc((size_t)G4 * 4);
  unsigned short* h_bf    = (unsigned short*)alloc((size_t)B_ * H_ * 2);
  float*          c_f     = (float*)         alloc((size_t)B_ * H_ * 4);
  unsigned short* pemb    = (unsigned short*)alloc((size_t)B_ * E_ * 2);
  unsigned short* z_bf    = (unsigned short*)alloc((size_t)B_ * KZ * 2);
  float*          gates   = (float*)         alloc((size_t)G4 * B_ * 4);
  float*          logits  = (float*)         alloc((size_t)O_ * B_ * 4);
  (void)ws_size; (void)in_sizes; (void)n_in; (void)out_size;

  // one-time (per launch) weight conversion + state init
  {
    int n;
    n = G4 * KZ; f2bf_kernel<<<(n + 255) / 256, 256, 0, stream>>>(W_ih,  Wih_bf,  n);
    n = G4 * H_; f2bf_kernel<<<(n + 255) / 256, 256, 0, stream>>>(W_hh,  Whh_bf,  n);
    n = O_ * H_; f2bf_kernel<<<(n + 255) / 256, 256, 0, stream>>>(W_lin, Wlin_bf, n);
    init_state_kernel<<<(B_ * H_) / 256, 256, 0, stream>>>(b_ih, b_hh, init_t,
                                                           bias_g, h_bf, c_f, pemb);
  }

  // sequential decode: stream ordering carries the step-to-step dependency
  for (int t = 0; t < T_; ++t) {
    zbuild_kernel<<<(B_ * KZ + 255) / 256, 256, 0, stream>>>(enc, pemb, z_bf, t);
    gates_kernel<<<G4 / 128, 128, 0, stream>>>(Wih_bf, Whh_bf, z_bf, h_bf, bias_g, gates);
    cell_kernel<<<(B_ * H_) / 256, 256, 0, stream>>>(gates, c_f, h_bf, seq_lens, t);
    logits_kernel<<<O_ / 128, 128, 0, stream>>>(Wlin_bf, h_bf, b_lin, logits);
    out_kernel<<<B_, 128, 0, stream>>>(logits, emb, seq_lens, pemb, out, t);
  }
}
